// LenetQ_70609262346489
// MI455X (gfx1250) — matrix operand order repaired and checked by reading, not compiled, not run on hardware
//
#include <hip/hip_runtime.h>

// ---------------------------------------------------------------------------
// Types for CDNA5 WMMA
// ---------------------------------------------------------------------------
typedef __attribute__((ext_vector_type(16))) _Float16 v16h;
typedef __attribute__((ext_vector_type(8)))  _Float16 v8h;
typedef __attribute__((ext_vector_type(8)))  float    v8f;
typedef __attribute__((ext_vector_type(4)))  float    v4f;
typedef __attribute__((ext_vector_type(4)))  int      v4i;

#define WMMA_F32_16x16x32_F16(A, B, C) \
    __builtin_amdgcn_wmma_f32_16x16x32_f16(false, (A), false, (B), (short)0, (C), false, false)

#if __has_builtin(__builtin_amdgcn_global_load_async_to_lds_b128)
#define HAVE_ASYNC_LDS 1
typedef __attribute__((address_space(1))) v4i* gv4i_ptr;   // global int4*
typedef __attribute__((address_space(3))) v4i* lv4i_ptr;   // LDS int4*
#endif

// ---------------------------------------------------------------------------
// Network dimensions
// ---------------------------------------------------------------------------
#define BATCH   8192
#define H1ELEM  (6 * 14 * 14)          // 1176 per image (f16), conv1 pooled out
#define K2      150                    // conv2 im2col K = 6*5*5
#define K2P     160                    // padded to 5*32
#define M2P     112                    // 100 conv2 positions padded to 7*16
#define K3      400                    // conv3 GEMM K = 16*5*5
#define K3P     416                    // 13*32
#define N3P     128                    // 120 -> 128
// fc1: K 120->128, N 84->96 ; fc2: K 84->96, N 10->16

__device__ __forceinline__ float aq(float x) {
    float c = fminf(fmaxf(x, 0.f), 1.f);
    return rintf(c * 3.f) * (1.f / 3.f);    // 2-bit DoReFa activation (RNE = jnp.round)
}

// ---------------------------------------------------------------------------
// WMMA fragment loaders — two aligned 128-bit loads each.
// A row-major [M][lda] (lda multiple of 8 halves, 16B-aligned base):
//   lane slice = A[m][k0+8*hf .. +7]  ++  A[m][k0+16+8*hf .. +7]
// B stored transposed BT[N][ldk] (K contiguous):
//   lane slice = BT[n][k0+8*hf .. +7]  ++  BT[n][k0+16+8*hf .. +7]
// ---------------------------------------------------------------------------
__device__ __forceinline__ v16h cat8(v8h lo, v8h hi) {
    return __builtin_shufflevector(lo, hi, 0, 1, 2, 3, 4, 5, 6, 7,
                                            8, 9, 10, 11, 12, 13, 14, 15);
}

__device__ __forceinline__ v16h load_A_frag(const _Float16* A, int lda, int m0, int k0, int lane) {
    const _Float16* p = A + (m0 + (lane & 15)) * lda + k0 + 8 * (lane >> 4);
    v8h lo = *(const v8h*)p;
    v8h hi = *(const v8h*)(p + 16);
    return cat8(lo, hi);
}

__device__ __forceinline__ v16h load_BT_frag(const _Float16* BT, int ldk, int k0, int n0, int lane) {
    const _Float16* p = BT + (n0 + (lane & 15)) * ldk + k0 + 8 * (lane >> 4);
    v8h lo = *(const v8h*)p;
    v8h hi = *(const v8h*)(p + 16);
    return cat8(lo, hi);
}
// C/D f32 16x16: element v of v8f -> M = v + 8*(lane>>4), N = lane&15.

// ---------------------------------------------------------------------------
// Small utility kernels
// ---------------------------------------------------------------------------
__global__ void k_zero(float* p, int n) {
    int i = blockIdx.x * blockDim.x + threadIdx.x;
    if (i < n) p[i] = 0.f;
}

__global__ void k_meanabs(const float* __restrict__ w, int n, float* out) {
    __shared__ float red[256];
    float s = 0.f;
    for (int i = threadIdx.x; i < n; i += 256) s += fabsf(w[i]);
    red[threadIdx.x] = s;
    __syncthreads();
#pragma unroll
    for (int o = 128; o > 0; o >>= 1) {
        if ((int)threadIdx.x < o) red[threadIdx.x] += red[threadIdx.x + o];
        __syncthreads();
    }
    if (threadIdx.x == 0) *out = red[0] / (float)n;
}

__device__ __forceinline__ float fsign(float w) {
    return (float)((w > 0.f) ? 1 : ((w < 0.f) ? -1 : 0));
}

// Binarize + pad weights into TRANSPOSED (N-major, K-contiguous) f16 B matrices.
__global__ void k_pack(const float* __restrict__ w2, const float* __restrict__ w3,
                       const float* __restrict__ wf1, const float* __restrict__ wf2,
                       const float* __restrict__ E,
                       _Float16* w2qT, _Float16* w3qT, _Float16* f1qT, _Float16* f2qT) {
    int t = blockIdx.x * blockDim.x + threadIdx.x;
    float E2 = E[0], E3 = E[1], E1 = E[2];
    if (t < 16 * K2P) {                      // w2qT [16][160]
        int n = t / K2P, k = t - n * K2P;
        float v = (k < K2) ? fsign(w2[n * K2 + k]) * E2 : 0.f;
        w2qT[t] = (_Float16)v;
    }
    if (t < N3P * K3P) {                     // w3qT [128][416]
        int n = t / K3P, k = t - n * K3P;
        float v = (n < 120 && k < K3) ? fsign(w3[n * K3 + k]) * E3 : 0.f;
        w3qT[t] = (_Float16)v;
    }
    if (t < 96 * 128) {                      // f1qT [96][128]
        int n = t >> 7, k = t & 127;
        float v = (n < 84 && k < 120) ? fsign(wf1[n * 120 + k]) * E1 : 0.f;
        f1qT[t] = (_Float16)v;
    }
    if (t < 16 * 96) {                       // f2qT [16][96] (fp32 head, f16-cast)
        int n = t / 96, k = t - n * 96;
        float v = (n < 10 && k < 84) ? wf2[n * 84 + k] : 0.f;
        f2qT[t] = (_Float16)v;
    }
}

__global__ void k_bn_finalize(const float* __restrict__ stats, const float* __restrict__ g,
                              const float* __restrict__ b, float count, int nch,
                              float* scale, float* shift) {
    int c = threadIdx.x;
    if (c < nch) {
        float m  = stats[2 * c] / count;
        float v  = stats[2 * c + 1] / count - m * m;
        float sc = g[c] * rsqrtf(v + 1e-5f);
        scale[c] = sc;
        shift[c] = b[c] - m * sc;
    }
}

// ---------------------------------------------------------------------------
// Block 1: fp32 direct conv (25 taps, memory-bound) — stats pass
// ---------------------------------------------------------------------------
__global__ void k_conv1_stats(const float* __restrict__ x, const float* __restrict__ w,
                              const float* __restrict__ bias, float* stats) {
    __shared__ float wsh[150], bsh[6], ssum[6], ssq[6];
    int tid = threadIdx.x;
    if (tid < 150) wsh[tid] = w[tid];
    if (tid < 6) { bsh[tid] = bias[tid]; ssum[tid] = 0.f; ssq[tid] = 0.f; }
    __syncthreads();
    int t  = blockIdx.x * 256 + tid;         // exactly 8192*784 threads
    int b_ = t / 784, r = t % 784;
    int oh = r / 28, ow = r % 28;
    const float* xp = x + b_ * 1024 + oh * 32 + ow;
    float patch[25];
#pragma unroll
    for (int i = 0; i < 5; ++i)
#pragma unroll
        for (int j = 0; j < 5; ++j) patch[i * 5 + j] = xp[i * 32 + j];
#pragma unroll
    for (int c = 0; c < 6; ++c) {
        float acc = bsh[c];
#pragma unroll
        for (int k = 0; k < 25; ++k) acc += patch[k] * wsh[c * 25 + k];
        atomicAdd(&ssum[c], acc);
        atomicAdd(&ssq[c], acc * acc);
    }
    __syncthreads();
    if (tid < 6) {
        atomicAdd(&stats[2 * tid],     ssum[tid]);
        atomicAdd(&stats[2 * tid + 1], ssq[tid]);
    }
}

// Block 1 apply: recompute conv1, BN+ReLU+2x2 maxpool, emit f16 [B][6][14][14]
__global__ void k_conv1_apply(const float* __restrict__ x, const float* __restrict__ w,
                              const float* __restrict__ bias, const float* __restrict__ scale,
                              const float* __restrict__ shift, _Float16* h1p) {
    __shared__ float wsh[150], bsh[6], scsh[6], shsh[6];
    int tid = threadIdx.x;
    if (tid < 150) wsh[tid] = w[tid];
    if (tid < 6) { bsh[tid] = bias[tid]; scsh[tid] = scale[tid]; shsh[tid] = shift[tid]; }
    __syncthreads();
    int t  = blockIdx.x * 256 + tid;         // exactly 8192*196 threads
    int b_ = t / 196, p = t % 196;
    int ph = p / 14, pw = p % 14;
    const float* xp = x + b_ * 1024 + (2 * ph) * 32 + 2 * pw;
    float patch[36];
#pragma unroll
    for (int i = 0; i < 6; ++i)
#pragma unroll
        for (int j = 0; j < 6; ++j) patch[i * 6 + j] = xp[i * 32 + j];
#pragma unroll
    for (int c = 0; c < 6; ++c) {
        float m = 0.f;                       // post-ReLU values are >= 0
#pragma unroll
        for (int dy = 0; dy < 2; ++dy)
#pragma unroll
            for (int dx = 0; dx < 2; ++dx) {
                float acc = bsh[c];
#pragma unroll
                for (int i = 0; i < 5; ++i)
#pragma unroll
                    for (int j = 0; j < 5; ++j)
                        acc += patch[(i + dy) * 6 + (j + dx)] * wsh[c * 25 + i * 5 + j];
                float y = fmaxf(acc * scsh[c] + shsh[c], 0.f);
                m = fmaxf(m, y);
            }
        h1p[b_ * H1ELEM + c * 196 + p] = (_Float16)m;
    }
}

// ---------------------------------------------------------------------------
// Block 2: WMMA im2col conv (M=100->112, K=150->160, N=16) per image.
// im2col tile is materialized in LDS so A-frags are 2x ds_load_b128.
// ---------------------------------------------------------------------------
template <bool STATS>
__global__ void k_conv2(const _Float16* __restrict__ h1p, const _Float16* __restrict__ w2qT,
                        const float* __restrict__ bias, const float* __restrict__ scale,
                        const float* __restrict__ shift, float* stats, _Float16* h2p) {
    __shared__ __align__(16) _Float16 img[H1ELEM];        // 2.3 KB
    __shared__ __align__(16) _Float16 Ai[M2P * K2P];      // 35.8 KB im2col tile
    __shared__ __align__(16) _Float16 wBT[16 * K2P];      // 5 KB
    __shared__ float outbuf[100 * 16];                    // 6.4 KB (apply mode)
    __shared__ float ssum[16], ssq[16];
    int tid = threadIdx.x, b = blockIdx.x;
    if (tid < 16) { ssum[tid] = 0.f; ssq[tid] = 0.f; }
    // wide staging: image (1176 = 147*8 halves) and transposed weights (320*8)
    for (int c = tid; c < 147; c += 256)
        ((v8h*)img)[c] = ((const v8h*)(h1p + (size_t)b * H1ELEM))[c];
    for (int c = tid; c < (16 * K2P) / 8; c += 256)
        ((v8h*)wBT)[c] = ((const v8h*)w2qT)[c];
    __syncthreads();
    // build im2col tile (rows >=100 and k >=150 zero-padded)
    for (int i = tid; i < M2P * K2P; i += 256) {
        int r = i / K2P, k = i - (i / K2P) * K2P;
        _Float16 val = (_Float16)0.f;
        if (r < 100 && k < K2) {
            int oh = r / 10, ow = r - (r / 10) * 10;
            int ci = k / 25, rem = k - ci * 25;
            int kh = rem / 5, kw = rem - kh * 5;
            val = img[ci * 196 + (oh + kh) * 14 + (ow + kw)];
        }
        Ai[i] = val;
    }
    __syncthreads();
    int w = tid >> 5, lane = tid & 31;
    v8f acc = {};
    if (w < 7) {                              // wave-uniform: EXEC all ones for WMMA
#pragma unroll
        for (int kt = 0; kt < 5; ++kt) {
            v16h a  = load_A_frag(Ai, K2P, w * 16, kt * 32, lane);
            v16h bb = load_BT_frag(wBT, K2P, kt * 32, 0, lane);
            acc = WMMA_F32_16x16x32_F16(a, bb, acc);
        }
    }
    int ch = lane & 15, hf = lane >> 4;
    if (STATS) {
        if (w < 7) {
            float bi = bias[ch], s = 0.f, q = 0.f;
#pragma unroll
            for (int v = 0; v < 8; ++v) {
                int m = w * 16 + v + 8 * hf;
                if (m < 100) { float y = acc[v] + bi; s += y; q += y * y; }
            }
            atomicAdd(&ssum[ch], s);
            atomicAdd(&ssq[ch], q);
        }
        __syncthreads();
        if (tid < 16) {
            atomicAdd(&stats[2 * tid],     ssum[tid]);
            atomicAdd(&stats[2 * tid + 1], ssq[tid]);
        }
    } else {
        if (w < 7) {
            float bi = bias[ch], sc = scale[ch], sh = shift[ch];
#pragma unroll
            for (int v = 0; v < 8; ++v) {
                int m = w * 16 + v + 8 * hf;
                if (m < 100) outbuf[m * 16 + ch] = aq((acc[v] + bi) * sc + sh);
            }
        }
        __syncthreads();
        for (int t = tid; t < 400; t += 256) {  // 2x2 maxpool over 10x10 -> 5x5
            int c = t & 15, p = t >> 4;
            int ph = p / 5, pw = p - ph * 5;
            int r = ph * 20 + pw * 2;
            float m0 = fmaxf(fmaxf(outbuf[r * 16 + c],        outbuf[(r + 1) * 16 + c]),
                             fmaxf(outbuf[(r + 10) * 16 + c], outbuf[(r + 11) * 16 + c]));
            h2p[(size_t)b * K3 + c * 25 + ph * 5 + pw] = (_Float16)m0;
        }
    }
}

// ---------------------------------------------------------------------------
// Block 3: WMMA GEMM [8192 x 400] * [400 x 120] (K->416, N->128), fused BN stats.
// A tile staged via gfx1250 async global->LDS (ASYNCcnt) when available.
// ---------------------------------------------------------------------------
__global__ void k_conv3(const _Float16* __restrict__ h2p, const _Float16* __restrict__ w3qT,
                        const float* __restrict__ bias, float* conv3_out, float* stats) {
    __shared__ __align__(16) _Float16 A[64 * K3P];        // 53 KB
    int tid = threadIdx.x, row0 = blockIdx.x * 64;
    __builtin_prefetch(&w3qT[tid * ((N3P * K3P) / 256)], 0, 3);  // keep B hot in L2
    // zero-fill the K-pad region (k = 400..415)
    for (int i = tid; i < 64 * 16; i += 256) {
        int r = i >> 4;
        A[r * K3P + K3 + (i & 15)] = (_Float16)0.f;
    }
    // bulk copy: 64 rows x 50 16-byte chunks (row stride 800B, 16B aligned)
    for (int i = tid; i < 64 * 50; i += 256) {
        int r = i / 50, c = i - r * 50;
#ifdef HAVE_ASYNC_LDS
        __builtin_amdgcn_global_load_async_to_lds_b128(
            (gv4i_ptr)(h2p + (size_t)(row0 + r) * K3 + c * 8),
            (lv4i_ptr)&A[r * K3P + c * 8], 0, 0);
#else
        *(v8h*)&A[r * K3P + c * 8] = *(const v8h*)(h2p + (size_t)(row0 + r) * K3 + c * 8);
#endif
    }
#ifdef HAVE_ASYNC_LDS
#if __has_builtin(__builtin_amdgcn_s_wait_asynccnt)
    __builtin_amdgcn_s_wait_asynccnt(0);
#else
    asm volatile("s_wait_asynccnt 0x0" ::: "memory");
#endif
#endif
    __syncthreads();
    int w = tid >> 5, lane = tid & 31;
    int mt = w & 3, ngrp = w >> 2;            // 8 waves: 4 M-tiles x 2 N-groups
    v8f acc[4] = {};
#pragma unroll
    for (int kt = 0; kt < 13; ++kt) {
        v16h a = load_A_frag(A, K3P, mt * 16, kt * 32, lane);   // reused 4x
#pragma unroll
        for (int nt = 0; nt < 4; ++nt) {
            v16h bb = load_BT_frag(w3qT, K3P, kt * 32, ngrp * 64 + nt * 16, lane);
            acc[nt] = WMMA_F32_16x16x32_F16(a, bb, acc[nt]);
        }
    }
#pragma unroll
    for (int nt = 0; nt < 4; ++nt) {
        int ch = ngrp * 64 + nt * 16 + (lane & 15);
        if (ch < 120) {
            float bi = bias[ch], s = 0.f, q = 0.f;
#pragma unroll
            for (int v = 0; v < 8; ++v) {
                int m = row0 + mt * 16 + v + 8 * (lane >> 4);
                float y = acc[nt][v] + bi;
                conv3_out[(size_t)m * 120 + ch] = y;
                s += y; q += y * y;
            }
            atomicAdd(&stats[2 * ch],     s);
            atomicAdd(&stats[2 * ch + 1], q);
        }
    }
}

// ---------------------------------------------------------------------------
// Head: bn3+aq -> fc1 (WMMA K=128,N=96) -> aq -> fc2 (WMMA K=96,N=16) -> out
// ---------------------------------------------------------------------------
__global__ void k_head(const float* __restrict__ conv3_out, const float* __restrict__ scale3,
                       const float* __restrict__ shift3, const _Float16* __restrict__ f1qT,
                       const float* __restrict__ fc1_b, const _Float16* __restrict__ f2qT,
                       const float* __restrict__ fc2_b, float* __restrict__ out) {
    __shared__ __align__(16) _Float16 A1[64 * 128];       // 16 KB
    __shared__ __align__(16) _Float16 A2[64 * 96];        // 12 KB
    int tid = threadIdx.x, row0 = blockIdx.x * 64;
    // stage bn3+aq activations: rows of 120 f32 -> 30 float4 chunks each
    for (int i = tid; i < 64 * 30; i += 256) {
        int r = i / 30, c4 = i - r * 30;
        v4f x4 = *(const v4f*)(conv3_out + (size_t)(row0 + r) * 120 + c4 * 4);
#pragma unroll
        for (int e = 0; e < 4; ++e) {
            int k = c4 * 4 + e;
            A1[r * 128 + k] = (_Float16)aq(x4[e] * scale3[k] + shift3[k]);
        }
    }
    for (int i = tid; i < 64 * 8; i += 256)               // K-pad 120..127
        A1[(i >> 3) * 128 + 120 + (i & 7)] = (_Float16)0.f;
    __syncthreads();
    int w = tid >> 5, lane = tid & 31;
    int mt = w & 3, ngrp = w >> 2;            // fc1: 4 M-tiles x 6 N-tiles over 8 waves
    {
        v8f acc[3] = {};
#pragma unroll
        for (int kt = 0; kt < 4; ++kt) {
            v16h a = load_A_frag(A1, 128, mt * 16, kt * 32, lane);   // reused 3x
#pragma unroll
            for (int j = 0; j < 3; ++j) {
                v16h bb = load_BT_frag(f1qT, 128, kt * 32, (ngrp * 3 + j) * 16, lane);
                acc[j] = WMMA_F32_16x16x32_F16(a, bb, acc[j]);
            }
        }
#pragma unroll
        for (int j = 0; j < 3; ++j) {
            int n = (ngrp * 3 + j) * 16 + (lane & 15);
            float bi = (n < 84) ? fc1_b[n] : 0.f;
#pragma unroll
            for (int v = 0; v < 8; ++v) {
                int m = mt * 16 + v + 8 * (lane >> 4);
                float q = (n < 84) ? aq(acc[j][v] + bi) : 0.f;
                A2[m * 96 + n] = (_Float16)q;
            }
        }
    }
    __syncthreads();
    if (w < 4) {                              // fc2: 4 M-tiles on waves 0..3
        v8f acc = {};
#pragma unroll
        for (int kt = 0; kt < 3; ++kt) {
            v16h a  = load_A_frag(A2, 96, w * 16, kt * 32, lane);
            v16h bb = load_BT_frag(f2qT, 96, kt * 32, 0, lane);
            acc = WMMA_F32_16x16x32_F16(a, bb, acc);
        }
        int n = lane & 15;
        if (n < 10) {
            float bi = fc2_b[n];
#pragma unroll
            for (int v = 0; v < 8; ++v) {
                int m = row0 + w * 16 + v + 8 * (lane >> 4);
                out[(size_t)m * 10 + n] = acc[v] + bi;
            }
        }
    }
}

// ---------------------------------------------------------------------------
// Host launcher
// ---------------------------------------------------------------------------
extern "C" void kernel_launch(void* const* d_in, const int* in_sizes, int n_in,
                              void* d_out, int out_size, void* d_ws, size_t ws_size,
                              hipStream_t stream) {
    (void)in_sizes; (void)n_in; (void)out_size; (void)ws_size;
    const float* x       = (const float*)d_in[0];
    const float* conv1_w = (const float*)d_in[1];
    const float* conv1_b = (const float*)d_in[2];
    const float* bn1_g   = (const float*)d_in[3];
    const float* bn1_b   = (const float*)d_in[4];
    const float* conv2_w = (const float*)d_in[5];
    const float* conv2_b = (const float*)d_in[6];
    const float* bn2_g   = (const float*)d_in[7];
    const float* bn2_b   = (const float*)d_in[8];
    const float* conv3_w = (const float*)d_in[9];
    const float* conv3_b = (const float*)d_in[10];
    const float* bn3_g   = (const float*)d_in[11];
    const float* bn3_b   = (const float*)d_in[12];
    const float* fc1_w   = (const float*)d_in[13];
    const float* fc1_b   = (const float*)d_in[14];
    const float* fc2_w   = (const float*)d_in[15];
    const float* fc2_b   = (const float*)d_in[16];
    float* out = (float*)d_out;

    // workspace carve-up (256B aligned)
    char* base = (char*)d_ws;
    size_t o = 0;
    auto carve = [&](size_t bytes) { char* p = base + o; o = (o + bytes + 255) & ~(size_t)255; return p; };
    _Float16* h1p   = (_Float16*)carve((size_t)BATCH * H1ELEM * 2);   // 19.3 MB
    _Float16* h2p   = (_Float16*)carve((size_t)BATCH * K3 * 2);       //  6.6 MB
    float*    c3out = (float*)   carve((size_t)BATCH * 120 * 4);      //  3.9 MB
    _Float16* w2qT  = (_Float16*)carve(16 * K2P * 2);
    _Float16* w3qT  = (_Float16*)carve(N3P * K3P * 2);
    _Float16* f1qT  = (_Float16*)carve(96 * 128 * 2);
    _Float16* f2qT  = (_Float16*)carve(16 * 96 * 2);
    float*    Ebuf  = (float*)   carve(3 * 4);
    float*    stats = (float*)   carve(284 * 4);   // [6 + 16 + 120] x {sum,sumsq}
    float* stats1 = stats, *stats2 = stats + 12, *stats3 = stats + 44;
    float* sc1 = (float*)carve(6 * 4),   *sh1 = (float*)carve(6 * 4);
    float* sc2 = (float*)carve(16 * 4),  *sh2 = (float*)carve(16 * 4);
    float* sc3 = (float*)carve(120 * 4), *sh3 = (float*)carve(120 * 4);

    // 0. zero the accumulators (determinism across graph replays)
    k_zero<<<1, 512, 0, stream>>>(stats, 284);
    // 1. weight-quant scales E = mean|w|
    k_meanabs<<<1, 256, 0, stream>>>(conv2_w, 2400,  Ebuf + 0);
    k_meanabs<<<1, 256, 0, stream>>>(conv3_w, 48000, Ebuf + 1);
    k_meanabs<<<1, 256, 0, stream>>>(fc1_w,   10080, Ebuf + 2);
    // 2. binarize + pad weights into transposed WMMA B layout
    k_pack<<<(N3P * K3P + 255) / 256, 256, 0, stream>>>(conv2_w, conv3_w, fc1_w, fc2_w,
                                                        Ebuf, w2qT, w3qT, f1qT, f2qT);
    // 3. block 1 (stats -> finalize -> apply; recompute instead of storing pre-BN)
    k_conv1_stats<<<(BATCH * 784) / 256, 256, 0, stream>>>(x, conv1_w, conv1_b, stats1);
    k_bn_finalize<<<1, 128, 0, stream>>>(stats1, bn1_g, bn1_b, 8192.f * 784.f, 6, sc1, sh1);
    k_conv1_apply<<<(BATCH * 196) / 256, 256, 0, stream>>>(x, conv1_w, conv1_b, sc1, sh1, h1p);
    // 4. block 2 (WMMA stats -> finalize -> WMMA apply + aq + pool)
    k_conv2<true ><<<BATCH, 256, 0, stream>>>(h1p, w2qT, conv2_b, sc2, sh2, stats2, h2p);
    k_bn_finalize<<<1, 128, 0, stream>>>(stats2, bn2_g, bn2_b, 8192.f * 100.f, 16, sc2, sh2);
    k_conv2<false><<<BATCH, 256, 0, stream>>>(h1p, w2qT, conv2_b, sc2, sh2, stats2, h2p);
    // 5. block 3 (single WMMA GEMM with fused stats)
    k_conv3<<<BATCH / 64, 256, 0, stream>>>(h2p, w3qT, conv3_b, c3out, stats3);
    k_bn_finalize<<<1, 128, 0, stream>>>(stats3, bn3_g, bn3_b, 8192.f, 120, sc3, sh3);
    // 6. fused head: bn3+aq -> fc1 -> aq -> fc2
    k_head<<<BATCH / 64, 256, 0, stream>>>(c3out, sc3, sh3, f1qT, fc1_b, f2qT, fc2_b, out);
}
